// MEGA_592705486882
// MI455X (gfx1250) — matrix operand order repaired and checked by reading, not compiled, not run on hardware
//
#include <hip/hip_runtime.h>
#include <stdint.h>

#define LL   2048
#define BB   8
#define SZ   512
#define NH   4
#define QKD  128
#define VD   1024
#define NBASE 2176   // SZ + QKD + VD + SZ

typedef __attribute__((ext_vector_type(16))) __bf16        bf16x16;
typedef __attribute__((ext_vector_type(8)))  float         f32x8;
typedef __attribute__((ext_vector_type(4)))  unsigned int  u32x4;

union FragU { u32x4 u[2]; bf16x16 v; };

#if defined(__gfx1250__) && __has_builtin(__builtin_amdgcn_global_load_async_to_lds_b128) && \
    __has_builtin(__builtin_amdgcn_s_wait_asynccnt)
#define HAVE_ASYNC_LDS 1
#else
#define HAVE_ASYNC_LDS 0
#endif

#if HAVE_ASYNC_LDS
typedef int v4i __attribute__((vector_size(16)));
typedef __attribute__((address_space(1))) v4i* gv4i_p;   // global ptr to int4
typedef __attribute__((address_space(3))) v4i* lv4i_p;   // LDS ptr to int4
#endif

__device__ __forceinline__ unsigned short f2bf(float f) {
  unsigned int x = __float_as_uint(f);
  x += 0x7fffu + ((x >> 16) & 1u);
  return (unsigned short)(x >> 16);
}
__device__ __forceinline__ float sigm(float x) { return 1.f / (1.f + __expf(-x)); }

// copy 16 bytes global -> LDS; OFF advances BOTH addresses (matches async IOFFSET semantics)
template <int OFF>
__device__ __forceinline__ void cp16_g2l(const unsigned short* g, unsigned short* l) {
#if HAVE_ASYNC_LDS
  __builtin_amdgcn_global_load_async_to_lds_b128((gv4i_p)g, (lv4i_p)l, OFF, 0);
#else
  *(u32x4*)((char*)l + OFF) = *(const u32x4*)((const char*)g + OFF);
#endif
}
__device__ __forceinline__ void wait_g2l() {
#if HAVE_ASYNC_LDS
  __builtin_amdgcn_s_wait_asynccnt(0);
#endif
}

// ---------------- f32 -> bf16 bulk convert ----------------
__global__ void cvt_bf16_kernel(const float* __restrict__ src,
                                unsigned short* __restrict__ dst, long n) {
  long i = (long)blockIdx.x * blockDim.x + threadIdx.x;
  long stride = (long)gridDim.x * blockDim.x;
  for (; i < n; i += stride) dst[i] = f2bf(src[i]);
}

// ---------------- multihead damped EMA as 4 IIR scans ----------------
__global__ __launch_bounds__(256) void ema_kernel(
    const float* __restrict__ x, const float* __restrict__ praw,
    const float* __restrict__ araw, const float* __restrict__ beta,
    const float* __restrict__ gamma, const float* __restrict__ omega,
    unsigned short* __restrict__ emab) {
  int idx = blockIdx.x * 256 + threadIdx.x;
  if (idx >= BB * SZ) return;
  int b = idx / SZ, d = idx % SZ;
  float q[NH], c[NH], s[NH];
#pragma unroll
  for (int h = 0; h < NH; ++h) {
    float p = sigm(praw[d * NH + h]);
    q[h] = 1.f - p * sigm(araw[d * NH + h]);
    c[h] = gamma[d * NH + h] * 0.5f * p * beta[d * NH + h];  // NH^-0.5 = 0.5
    s[h] = 0.f;
  }
  float om = omega[d];
  for (int l = 0; l < LL; ++l) {
    float xv = x[((long)l * BB + b) * SZ + d];
    float a = xv * om;
#pragma unroll
    for (int h = 0; h < NH; ++h) { s[h] = q[h] * s[h] + xv; a += c[h] * s[h]; }
    emab[((long)l * BB + b) * SZ + d] = f2bf(fmaxf(a, 0.f));
  }
}

// ---------------- one-shot transpose: value (L*B, V) -> valueT (B, V, L) ----------------
__global__ __launch_bounds__(256) void transpose_v_kernel(
    const unsigned short* __restrict__ src, unsigned short* __restrict__ dst) {
  __shared__ unsigned short tile[32][33];
  int b  = blockIdx.z;
  int l0 = blockIdx.x * 32;
  int v0 = blockIdx.y * 32;
  int tx = threadIdx.x & 31;
  int ty = threadIdx.x >> 5;  // 0..7
#pragma unroll
  for (int j = 0; j < 4; ++j) {
    int l = l0 + ty + j * 8;
    tile[ty + j * 8][tx] = src[((long)l * BB + b) * VD + v0 + tx];
  }
  __syncthreads();
#pragma unroll
  for (int j = 0; j < 4; ++j) {
    int v = v0 + ty + j * 8;
    dst[((long)b * VD + v) * LL + l0 + tx] = tile[tx][ty + j * 8];
  }
}

// ---------------- generic 128x128x32 bf16 WMMA GEMM (double-buffered LDS) ----------------
// A: MxK row-major. Bm: NxK row-major (operand columns = rows of Bm).
// MODE 0: value  = relu(A@W^T + bias)            -> bf16
// MODE 1: base   = A@W^T + bias                  -> f32
// MODE 2: scores = A@W^T + relw[2047+n-m]        -> f32 (per-batch z)
// MODE 3: hin    = (P@V) * relu(gate)            -> bf16, row remap m*BB+z
// MODE 4: h      = relu(A@W^T + bias + extra)    -> f32
template <int MODE>
__global__ __launch_bounds__(256) void gemm_wmma_kernel(
    const unsigned short* __restrict__ A, int lda, long aStride,
    const unsigned short* __restrict__ Bm, int ldb, long bStride,
    const float* __restrict__ bias,
    const float* __restrict__ extra, int extraLd,
    const float* __restrict__ relw,
    const float* __restrict__ gateBase,
    float* __restrict__ Cf, long cStride,
    unsigned short* __restrict__ Cb,
    int M, int N, int K) {
  __shared__ __align__(16) unsigned short As[2][128][40];
  __shared__ __align__(16) unsigned short Bs[2][128][40];

  const int z = blockIdx.z;
  const unsigned short* Ap = A + (long)z * aStride;
  const unsigned short* Bp = Bm + (long)z * bStride;

  const int n0 = blockIdx.x * 128;
  const int m0 = blockIdx.y * 128;

  const int t    = threadIdx.x;
  const int wave = t >> 5;
  const int lane = t & 31;
  const int wm   = wave >> 1;  // 0..3 -> M offset wm*32
  const int wn   = wave & 1;   // 0..1 -> N offset wn*64
  const int l15  = lane & 15;
  const bool lo  = lane < 16;

  const int arow  = t >> 1;          // 0..127
  const int ahalf = (t & 1) << 4;    // 0 or 16

  f32x8 acc[2][4];
#pragma unroll
  for (int i = 0; i < 2; ++i)
#pragma unroll
    for (int j = 0; j < 4; ++j) acc[i][j] = (f32x8){0, 0, 0, 0, 0, 0, 0, 0};

  auto load_tiles = [&](int k0, int buf) {
    const unsigned short* ga = Ap + (long)(m0 + arow) * lda + k0 + ahalf;
    unsigned short* la = &As[buf][arow][ahalf];
    cp16_g2l<0>(ga, la);
    cp16_g2l<16>(ga, la);
    const unsigned short* gb = Bp + (long)(n0 + arow) * ldb + k0 + ahalf;
    unsigned short* lb = &Bs[buf][arow][ahalf];
    cp16_g2l<0>(gb, lb);
    cp16_g2l<16>(gb, lb);
  };

  const int nk = K >> 5;
  load_tiles(0, 0);
  wait_g2l();
  __syncthreads();

  for (int ki = 0; ki < nk; ++ki) {
    const int cur = ki & 1;
    if (ki + 1 < nk) load_tiles((ki + 1) << 5, cur ^ 1);

    // A fragment: lanes 0-15 hold K{0..7,16..23}; lanes 16-31 hold K{8..15,24..31}
    FragU af[2];
#pragma unroll
    for (int mt = 0; mt < 2; ++mt) {
      int r  = wm * 32 + mt * 16 + l15;
      int kb = lo ? 0 : 8;
      af[mt].u[0] = *(const u32x4*)&As[cur][r][kb];
      af[mt].u[1] = *(const u32x4*)&As[cur][r][kb + 16];
    }
    // B fragment: lanes 0-15 hold K{0..7,16..23}; lanes 16-31 hold K{8..15,24..31}
    FragU bfr[4];
#pragma unroll
    for (int nt = 0; nt < 4; ++nt) {
      int c  = wn * 64 + nt * 16 + l15;
      int kb = lo ? 0 : 8;
      bfr[nt].u[0] = *(const u32x4*)&Bs[cur][c][kb];
      bfr[nt].u[1] = *(const u32x4*)&Bs[cur][c][kb + 16];
    }
#pragma unroll
    for (int mt = 0; mt < 2; ++mt)
#pragma unroll
      for (int nt = 0; nt < 4; ++nt)
        acc[mt][nt] = __builtin_amdgcn_wmma_f32_16x16x32_bf16(
            false, af[mt].v, false, bfr[nt].v, (short)0, acc[mt][nt], false, false);

    wait_g2l();
    __syncthreads();
  }

  // Epilogue. C layout: vgpr r -> row = r + (lane<16 ? 0 : 8), col = lane&15
#pragma unroll
  for (int mt = 0; mt < 2; ++mt)
#pragma unroll
    for (int nt = 0; nt < 4; ++nt)
#pragma unroll
      for (int r = 0; r < 8; ++r) {
        int rl = wm * 32 + mt * 16 + r + (lo ? 0 : 8);
        int cl = wn * 64 + nt * 16 + l15;
        long m = m0 + rl;
        long n = n0 + cl;
        float v = acc[mt][nt][r];
        if constexpr (MODE == 0) {
          v = fmaxf(v + bias[n], 0.f);
          Cb[m * (long)N + n] = f2bf(v);
        } else if constexpr (MODE == 1) {
          Cf[m * (long)N + n] = v + bias[n];
        } else if constexpr (MODE == 2) {
          v += relw[2047 + (int)n - (int)m];
          (Cf + (long)z * cStride)[m * (long)N + n] = v;
        } else if constexpr (MODE == 3) {
          long row = m * BB + z;
          float g = fmaxf(gateBase[row * (long)NBASE + (SZ + QKD) + n], 0.f);
          Cb[row * (long)N + n] = f2bf(v * g);
        } else {  // MODE 4
          v += bias[n] + extra[m * (long)extraLd + n];
          Cf[m * (long)N + n] = fmaxf(v, 0.f);
        }
      }
  (void)M;
}

// ---------------- qk -> scaled query / key (bf16, (b,l,k) layout) ----------------
__global__ void makeqk_kernel(const float* __restrict__ base,
                              const float* __restrict__ qkw,
                              const float* __restrict__ qkb,
                              unsigned short* __restrict__ qb,
                              unsigned short* __restrict__ kb) {
  long i = (long)blockIdx.x * 256 + threadIdx.x;
  if (i >= (long)LL * BB * QKD) return;
  long m = i >> 7;
  int k = (int)(i & 127);
  long l = m >> 3;
  int b = (int)(m & 7);
  float qv = fmaxf(base[m * (long)NBASE + SZ + k], 0.f);
  float qq = (qv * qkw[k] + qkb[k]) * 0.0883883476483184f;  // QK^-0.5 folded in
  float kk = qv * qkw[QKD + k] + qkb[QKD + k];
  long o = ((long)b * LL + l) * QKD + k;
  qb[o] = f2bf(qq);
  kb[o] = f2bf(kk);
}

// ---------------- softmax over rows of 2048, write bf16 probs in place ----------------
__global__ __launch_bounds__(256) void softmax_kernel(float* __restrict__ scores) {
  __shared__ float red[256];
  long row = blockIdx.x;
  const float* s = scores + row * (long)LL;
  // bf16 probs overlay this row's own fp32 storage (row stride 4096 ushorts)
  unsigned short* p = (unsigned short*)scores + row * (long)(2 * LL);
  float v[8];
  float mx = -3.4e38f;
#pragma unroll
  for (int j = 0; j < 8; ++j) {
    v[j] = s[threadIdx.x + 256 * j];
    mx = fmaxf(mx, v[j]);
  }
  red[threadIdx.x] = mx; __syncthreads();
  for (int st = 128; st > 0; st >>= 1) {
    if (threadIdx.x < st) red[threadIdx.x] = fmaxf(red[threadIdx.x], red[threadIdx.x + st]);
    __syncthreads();
  }
  mx = red[0]; __syncthreads();
  float sum = 0.f;
#pragma unroll
  for (int j = 0; j < 8; ++j) { v[j] = __expf(v[j] - mx); sum += v[j]; }
  red[threadIdx.x] = sum; __syncthreads();
  for (int st = 128; st > 0; st >>= 1) {
    if (threadIdx.x < st) red[threadIdx.x] += red[threadIdx.x + st];
    __syncthreads();
  }
  float inv = 1.f / red[0];
  __syncthreads();
#pragma unroll
  for (int j = 0; j < 8; ++j) p[threadIdx.x + 256 * j] = f2bf(v[j] * inv);
}

// ---------------- gated residual + LayerNorm ----------------
__global__ __launch_bounds__(256) void ln_kernel(
    const float* __restrict__ h, const float* __restrict__ x,
    const float* __restrict__ base, const float* __restrict__ lnw,
    const float* __restrict__ lnb, float* __restrict__ out) {
  __shared__ float red[256];
  long m = blockIdx.x;
  float v[2];
#pragma unroll
  for (int j = 0; j < 2; ++j) {
    int n = threadIdx.x + 256 * j;
    float rw = sigm(base[m * (long)NBASE + n]);
    float hv = h[m * (long)SZ + n];
    float xv = x[m * (long)SZ + n];
    v[j] = xv + rw * (hv - xv);
  }
  red[threadIdx.x] = v[0] + v[1]; __syncthreads();
  for (int st = 128; st > 0; st >>= 1) {
    if (threadIdx.x < st) red[threadIdx.x] += red[threadIdx.x + st];
    __syncthreads();
  }
  float mu = red[0] * (1.f / 512.f); __syncthreads();
  red[threadIdx.x] = (v[0] - mu) * (v[0] - mu) + (v[1] - mu) * (v[1] - mu);
  __syncthreads();
  for (int st = 128; st > 0; st >>= 1) {
    if (threadIdx.x < st) red[threadIdx.x] += red[threadIdx.x + st];
    __syncthreads();
  }
  float inv = rsqrtf(red[0] * (1.f / 512.f) + 1e-5f);
  __syncthreads();
#pragma unroll
  for (int j = 0; j < 2; ++j) {
    int n = threadIdx.x + 256 * j;
    out[m * (long)SZ + n] = (v[j] - mu) * inv * lnw[n] + lnb[n];
  }
}

extern "C" void kernel_launch(void* const* d_in, const int* in_sizes, int n_in,
                              void* d_out, int out_size, void* d_ws, size_t ws_size,
                              hipStream_t stream) {
  const float* x    = (const float*)d_in[0];
  const float* pvw  = (const float*)d_in[1];
  const float* pvb  = (const float*)d_in[2];
  const float* pmxw = (const float*)d_in[3];
  const float* pmxb = (const float*)d_in[4];
  const float* phw  = (const float*)d_in[5];
  const float* phb  = (const float*)d_in[6];
  const float* qkw  = (const float*)d_in[7];
  const float* qkb  = (const float*)d_in[8];
  const float* damp = (const float*)d_in[9];
  const float* deca = (const float*)d_in[10];
  const float* emae = (const float*)d_in[11];
  const float* emak = (const float*)d_in[12];
  const float* emar = (const float*)d_in[13];
  const float* relw = (const float*)d_in[14];
  const float* lnw  = (const float*)d_in[15];
  const float* lnb  = (const float*)d_in[16];
  float* out = (float*)d_out;

  char* ws = (char*)d_ws;
  size_t off = 0;
  auto alloc = [&](size_t bytes) {
    char* p = ws + off;
    off = (off + bytes + 255) & ~(size_t)255;
    return p;
  };
  unsigned short* xb    = (unsigned short*)alloc((size_t)LL * BB * SZ * 2);
  unsigned short* emab  = (unsigned short*)alloc((size_t)LL * BB * SZ * 2);
  unsigned short* wvb   = (unsigned short*)alloc((size_t)VD * SZ * 2);
  unsigned short* wmxb  = (unsigned short*)alloc((size_t)NBASE * SZ * 2);
  unsigned short* whb   = (unsigned short*)alloc((size_t)SZ * VD * 2);
  unsigned short* value = (unsigned short*)alloc((size_t)LL * BB * VD * 2);
  unsigned short* valT  = (unsigned short*)alloc((size_t)BB * VD * LL * 2);
  float*          basep = (float*)alloc((size_t)LL * BB * NBASE * 4);
  unsigned short* qb    = (unsigned short*)alloc((size_t)BB * LL * QKD * 2);
  unsigned short* kbuf  = (unsigned short*)alloc((size_t)BB * LL * QKD * 2);
  float*          scors = (float*)alloc((size_t)BB * LL * LL * 4);
  unsigned short* hin   = (unsigned short*)alloc((size_t)LL * BB * VD * 2);
  float*          hbuf  = (float*)alloc((size_t)LL * BB * SZ * 4);
  (void)ws_size; (void)in_sizes; (void)n_in; (void)out_size;

  // bf16 copies of activations/weights for WMMA
  cvt_bf16_kernel<<<4096, 256, 0, stream>>>(x, xb, (long)LL * BB * SZ);
  cvt_bf16_kernel<<<512, 256, 0, stream>>>(pvw, wvb, (long)VD * SZ);
  cvt_bf16_kernel<<<512, 256, 0, stream>>>(pmxw, wmxb, (long)NBASE * SZ);
  cvt_bf16_kernel<<<512, 256, 0, stream>>>(phw, whb, (long)SZ * VD);

  // EMA: 4 IIR scans per (b,d), relu(+residual) fused, bf16 out
  ema_kernel<<<(BB * SZ + 255) / 256, 256, 0, stream>>>(x, damp, deca, emae, emak, emar, emab);

  // value = relu(x @ Wv^T + bv) -> bf16 (L*B, V)
  gemm_wmma_kernel<0><<<dim3(VD / 128, (LL * BB) / 128, 1), 256, 0, stream>>>(
      xb, SZ, 0, wvb, SZ, 0, pvb, nullptr, 0, nullptr, nullptr,
      nullptr, 0, value, LL * BB, VD, SZ);

  // valueT (B, V, L) so P@V uses the fast NxK operand path
  transpose_v_kernel<<<dim3(LL / 32, VD / 32, BB), 256, 0, stream>>>(value, valT);

  // base = ema @ Wmx^T + bmx -> f32 (L*B, 2176)
  gemm_wmma_kernel<1><<<dim3(NBASE / 128, (LL * BB) / 128, 1), 256, 0, stream>>>(
      emab, SZ, 0, wmxb, SZ, 0, pmxb, nullptr, 0, nullptr, nullptr,
      basep, 0, nullptr, LL * BB, NBASE, SZ);

  makeqk_kernel<<<(unsigned)(((long)LL * BB * QKD + 255) / 256), 256, 0, stream>>>(
      basep, qkw, qkb, qb, kbuf);

  // scores[b] = Q K^T + relpos bias -> f32 (8, 2048, 2048)
  gemm_wmma_kernel<2><<<dim3(LL / 128, LL / 128, BB), 256, 0, stream>>>(
      qb, QKD, (long)LL * QKD, kbuf, QKD, (long)LL * QKD, nullptr, nullptr, 0, relw,
      nullptr, scors, (long)LL * LL, nullptr, LL, LL, QKD);

  // softmax rows, bf16 probs overlaid in scores buffer (row stride 4096 ushorts)
  softmax_kernel<<<BB * LL, 256, 0, stream>>>(scors);

  // hin = (P @ V) * relu(att_gate) -> bf16 (L*B, V)
  gemm_wmma_kernel<3><<<dim3(VD / 128, LL / 128, BB), 256, 0, stream>>>(
      (unsigned short*)scors, 2 * LL, (long)LL * 2 * LL, valT, LL, (long)VD * LL,
      nullptr, nullptr, 0, nullptr, basep, nullptr, 0, hin, LL, VD, LL);

  // h = relu(intermediate + hin @ Wh^T + bh) -> f32 (L*B, 512)
  gemm_wmma_kernel<4><<<dim3(SZ / 128, (LL * BB) / 128, 1), 256, 0, stream>>>(
      hin, VD, 0, whb, VD, 0, phb, basep + (SZ + QKD + VD), NBASE, nullptr, nullptr,
      hbuf, 0, nullptr, LL * BB, SZ, VD);

  // gated residual + LayerNorm
  ln_kernel<<<LL * BB, 256, 0, stream>>>(hbuf, x, basep, lnw, lnb, out);
}
